// Block_56891136803005
// MI455X (gfx1250) — matrix operand order repaired and checked by reading, not compiled, not run on hardware
//
#include <hip/hip_runtime.h>
#include <hip/hip_bf16.h>

// ---------------------------------------------------------------------------
// Types / helpers
// ---------------------------------------------------------------------------
typedef __attribute__((ext_vector_type(16))) __bf16 v16bf;
typedef __attribute__((ext_vector_type(8)))  __bf16 v8bf;
typedef __attribute__((ext_vector_type(8)))  float  v8f;

typedef unsigned short bf16; // raw bf16 storage

union V16 { v16bf v; v8bf h[2]; };

__device__ __forceinline__ bf16 f32_to_bf16(float f) {
    union { float f; unsigned int u; } c; c.f = f;
    unsigned int u = c.u;
    return (bf16)((u + 0x7FFFu + ((u >> 16) & 1u)) >> 16);
}
__device__ __forceinline__ float bf16_to_f32(bf16 h) {
    union { float f; unsigned int u; } c; c.u = ((unsigned int)h) << 16;
    return c.f;
}
__device__ __forceinline__ v8f wmma_v(v16bf a, v16bf b, v8f c) {
    return __builtin_amdgcn_wmma_f32_16x16x32_bf16(
        /*neg_a=*/false, a, /*neg_b=*/false, b,
        /*c_mod=*/(short)0, c, /*reuse_a=*/false, /*reuse_b=*/false);
}
__device__ __forceinline__ unsigned lds_addr(const void* p) {
    return (unsigned)(uintptr_t)p; // low 32 bits of flat shared addr == LDS offset
}

// Problem constants
#define BB 4
#define TT 2048
#define DD 1024
#define HH 16
#define HS 64
#define NGRP 2
#define PGRP 1024
#define SEQ 1025          // PGRP + 1 (summary token)
#define NEGV (-1e30f)

// ---------------------------------------------------------------------------
// Weight cast + transpose: W[K][N] f32 -> Wt[N][K] bf16 (K-major for B-frags)
// ---------------------------------------------------------------------------
__global__ __launch_bounds__(256)
void transpose_cast(const float* __restrict__ W, bf16* __restrict__ Wt,
                    int K, int N) {
    __shared__ float tile[32][33];
    const int n0 = blockIdx.x * 32, k0 = blockIdx.y * 32;
    const int tx = threadIdx.x & 31, ty = threadIdx.x >> 5; // ty 0..7
#pragma unroll
    for (int i = 0; i < 4; i++) {
        int k = ty + i * 8;
        tile[k][tx] = W[(size_t)(k0 + k) * N + n0 + tx];
    }
    __syncthreads();
#pragma unroll
    for (int i = 0; i < 4; i++) {
        int n = ty + i * 8;
        Wt[(size_t)(n0 + n) * K + k0 + tx] = f32_to_bf16(tile[tx][n]);
    }
}

// ---------------------------------------------------------------------------
// RMSNorm: one block per row (D = 1024), bf16 output
// ---------------------------------------------------------------------------
__global__ __launch_bounds__(256)
void rmsnorm_bf16(const float* __restrict__ x, const float* __restrict__ w,
                  bf16* __restrict__ out) {
    const int row = blockIdx.x;
    const float* xr = x + (size_t)row * DD;
    float s = 0.f;
    for (int i = threadIdx.x; i < DD; i += 256) { float v = xr[i]; s += v * v; }
    __shared__ float red[256];
    red[threadIdx.x] = s;
    __syncthreads();
    for (int st = 128; st > 0; st >>= 1) {
        if (threadIdx.x < st) red[threadIdx.x] += red[threadIdx.x + st];
        __syncthreads();
    }
    const float scale = rsqrtf(red[0] * (1.0f / DD) + 1e-5f);
    for (int i = threadIdx.x; i < DD; i += 256)
        out[(size_t)row * DD + i] = f32_to_bf16(xr[i] * scale * w[i]);
}

// ---------------------------------------------------------------------------
// WMMA GEMM: C[M][N] = A[M][K] @ Bt[N][K]^T  (A bf16 row-major, Bt bf16 K-major)
// Wave tile 16x64 (4 accumulators), 8 waves -> block tile 128x64, KSTEP 64.
// B tile (64x64 bf16, 8 KB) staged in LDS via GLOBAL_LOAD_ASYNC_TO_LDS_B128
// (shared by all 8 waves); A fragments streamed from global.
// EPI (compile-time): 0 = bf16 out, 1 = bf16 out + bias,
//                     2 = f32 out + residual, 3 = gated SwiGLU bf16 out
// ---------------------------------------------------------------------------
template <int EPI>
__global__ __launch_bounds__(256)
void gemm_wmma(const bf16* __restrict__ A, const bf16* __restrict__ Bt,
               const bf16* __restrict__ Bt2, const float* __restrict__ bias,
               const float* __restrict__ resid, float* __restrict__ outF,
               bf16* __restrict__ outB, int M, int N, int K) {
    __shared__ __align__(32) bf16 tileB[64 * 64];
    __shared__ __align__(32) bf16 tileB2[(EPI == 3) ? 64 * 64 : 32];

    const int tid = threadIdx.x;
    const int wave = tid >> 5;
    const int lane = tid & 31;
    const int lm = lane & 15, lh = lane >> 4;
    const int row0 = blockIdx.y * 128 + wave * 16;
    const int col0 = blockIdx.x * 64;

    // B staging: 512 16-byte chunks; 2 per thread (cols 0..31 and 32..63)
    const int bcol = tid >> 3;           // 0..31
    const int bkof = (tid & 7) * 8;      // 0,8,...,56
    const bf16* bsrc0 = Bt + (size_t)(col0 + bcol) * K + bkof;
    const bf16* bsrc1 = Bt + (size_t)(col0 + 32 + bcol) * K + bkof;
    const unsigned ldsb0 = lds_addr(&tileB[bcol * 64 + bkof]);
    const unsigned ldsb1 = lds_addr(&tileB[(32 + bcol) * 64 + bkof]);
    const bf16* b2src0 = (EPI == 3) ? Bt2 + (size_t)(col0 + bcol) * K + bkof : nullptr;
    const bf16* b2src1 = (EPI == 3) ? Bt2 + (size_t)(col0 + 32 + bcol) * K + bkof : nullptr;
    const unsigned lds20 = lds_addr(&tileB2[0]) + (unsigned)((bcol * 64 + bkof) * 2);
    const unsigned lds21 = lds_addr(&tileB2[0]) + (unsigned)(((32 + bcol) * 64 + bkof) * 2);

    const bf16* arow = A + (size_t)(row0 + lm) * K;

    v8f acc[4] = {};
    v8f acc2[4] = {};

    for (int k0 = 0; k0 < K; k0 += 64) {
        // async stage B tile(s) into LDS (ASYNCcnt path)
        asm volatile("global_load_async_to_lds_b128 %0, %1, off"
                     :: "v"(ldsb0), "v"(bsrc0 + k0) : "memory");
        asm volatile("global_load_async_to_lds_b128 %0, %1, off"
                     :: "v"(ldsb1), "v"(bsrc1 + k0) : "memory");
        if (EPI == 3) {
            asm volatile("global_load_async_to_lds_b128 %0, %1, off"
                         :: "v"(lds20), "v"(b2src0 + k0) : "memory");
            asm volatile("global_load_async_to_lds_b128 %0, %1, off"
                         :: "v"(lds21), "v"(b2src1 + k0) : "memory");
        }

        // A fragments for K sub-steps k0..k0+31 and k0+32..k0+63 (A-layout)
        const v8bf* ap = reinterpret_cast<const v8bf*>(arow + k0);
        V16 a0, a1;
        a0.h[0] = ap[2 * lh];     a0.h[1] = ap[2 * lh + 1];
        a1.h[0] = ap[4 + 2 * lh]; a1.h[1] = ap[5 + 2 * lh];

        asm volatile("s_wait_asynccnt 0x0" ::: "memory");
        __syncthreads();

#pragma unroll
        for (int j = 0; j < 4; j++) {
            const v16bf* bp =
                reinterpret_cast<const v16bf*>(&tileB[(j * 16 + lm) * 64]);
            acc[j] = wmma_v(a0.v, bp[lh], acc[j]);       // K = k0    + lh*16 + j'
            acc[j] = wmma_v(a1.v, bp[2 + lh], acc[j]);   // K = k0+32 + lh*16 + j'
            if (EPI == 3) {
                const v16bf* b2p =
                    reinterpret_cast<const v16bf*>(&tileB2[(j * 16 + lm) * 64]);
                acc2[j] = wmma_v(a0.v, b2p[lh], acc2[j]);
                acc2[j] = wmma_v(a1.v, b2p[2 + lh], acc2[j]);
            }
        }
        asm volatile("s_wait_dscnt 0x0" ::: "memory");
        __syncthreads();
    }

#pragma unroll
    for (int j = 0; j < 4; j++) {
        const int col = col0 + j * 16 + lm;
        float bv = 0.f;
        if (EPI == 1) bv = bias[col];
#pragma unroll
        for (int r = 0; r < 8; r++) {
            const int row = row0 + r + 8 * lh;
            const size_t idx = (size_t)row * N + col;
            const float u = acc[j][r] + bv;
            if (EPI == 3) {
                const float g = acc2[j][r];
                const float sg = u / (1.f + __expf(-u)); // silu
                outB[idx] = f32_to_bf16(sg * g);
            } else if (EPI == 2) {
                outF[idx] = resid[idx] + u;
            } else {
                outB[idx] = f32_to_bf16(u);
            }
        }
    }
}

// ---------------------------------------------------------------------------
// Pack Q/K/V into (B,H,NG,SEQ,HS) bf16 with appended mean summary row.
// ---------------------------------------------------------------------------
__global__ __launch_bounds__(256)
void pack_qkv(const bf16* __restrict__ qkbf, const bf16* __restrict__ vbf,
              bf16* __restrict__ Qg, bf16* __restrict__ Kg, bf16* __restrict__ Vg,
              float* __restrict__ qmean, float* __restrict__ kmean) {
    const int bhg = blockIdx.x;               // (b*H + h)*NG + g
    const int g = bhg % NGRP;
    const int bh = bhg / NGRP;
    const int h = bh % HH, b = bh / HH;
    const int hs = threadIdx.x & 63;
    const int pc = threadIdx.x >> 6;          // 0..3

    float sq = 0.f, sk = 0.f, sv = 0.f;
    for (int i = 0; i < PGRP / 4; i++) {
        const int p = pc * (PGRP / 4) + i;
        const int t = g * PGRP + p;
        const size_t rq = ((size_t)(b * TT + t)) * (2 * DD) + h * HS + hs;
        const size_t rv = ((size_t)(b * TT + t)) * DD + h * HS + hs;
        const bf16 q = qkbf[rq];
        const bf16 k = qkbf[rq + DD];
        const bf16 v = vbf[rv];
        const size_t o = ((size_t)bhg * SEQ + p) * HS + hs;
        Qg[o] = q; Kg[o] = k; Vg[o] = v;
        sq += bf16_to_f32(q); sk += bf16_to_f32(k); sv += bf16_to_f32(v);
    }
    __shared__ float red[3][4][64];
    red[0][pc][hs] = sq; red[1][pc][hs] = sk; red[2][pc][hs] = sv;
    __syncthreads();
    if (pc == 0) {
        const float inv = 1.f / (float)PGRP;
        const float mq = (red[0][0][hs] + red[0][1][hs] + red[0][2][hs] + red[0][3][hs]) * inv;
        const float mk = (red[1][0][hs] + red[1][1][hs] + red[1][2][hs] + red[1][3][hs]) * inv;
        const float mv = (red[2][0][hs] + red[2][1][hs] + red[2][2][hs] + red[2][3][hs]) * inv;
        const size_t o = ((size_t)bhg * SEQ + PGRP) * HS + hs;
        Qg[o] = f32_to_bf16(mq);
        Kg[o] = f32_to_bf16(mk);
        Vg[o] = f32_to_bf16(mv);
        qmean[(size_t)bhg * HS + hs] = mq;
        kmean[(size_t)bhg * HS + hs] = mk;
    }
}

// ---------------------------------------------------------------------------
// Flash attention (causal), one wave per 16-query tile of one (b,h,g).
// QK^T via 4 WMMA, online softmax (shfl_xor in 16-lane halves), P restaged
// through LDS into A-layout, V staged transposed in LDS, O += P@V via 4 WMMA.
// ---------------------------------------------------------------------------
__global__ __launch_bounds__(32)
void attn_wmma(const bf16* __restrict__ Qg, const bf16* __restrict__ Kg,
               const bf16* __restrict__ Vg, bf16* __restrict__ attout,
               float* __restrict__ attsum) {
    const int lane = threadIdx.x & 31;
    const int lm = lane & 15, lh = lane >> 4;
    const int bhg = blockIdx.y;
    const int g = bhg & (NGRP - 1);
    const int bh = bhg >> 1;
    const int h = bh & (HH - 1), b = bh >> 4;
    const int q0 = blockIdx.x * 16;

    const bf16* Q = Qg + (size_t)bhg * SEQ * HS;
    const bf16* Kp = Kg + (size_t)bhg * SEQ * HS;
    const bf16* Vp = Vg + (size_t)bhg * SEQ * HS;

    __shared__ __align__(32) bf16 Pl[16 * 32];
    __shared__ __align__(32) bf16 Vt[64 * 32];

    // Q fragments (A-layout), d 0..31 and 32..63
    int qrow = q0 + lm; if (qrow > SEQ - 1) qrow = SEQ - 1;
    const v8bf* qp = reinterpret_cast<const v8bf*>(Q + (size_t)qrow * HS);
    V16 qa0, qa1;
    qa0.h[0] = qp[2 * lh];     qa0.h[1] = qp[2 * lh + 1];
    qa1.h[0] = qp[4 + 2 * lh]; qa1.h[1] = qp[5 + 2 * lh];

    float mrow[8], lrow[8];
#pragma unroll
    for (int r = 0; r < 8; r++) { mrow[r] = NEGV; lrow[r] = 0.f; }
    v8f o[4] = {};

    int lastkey = q0 + 15; if (lastkey > SEQ - 1) lastkey = SEQ - 1;
    const int nkt = lastkey / 32 + 1;
    const float scale = 0.125f; // 1/sqrt(64)

    for (int kt = 0; kt < nkt; kt++) {
        const int k0 = kt * 32;
        const int key0 = k0 + lm;
        const int key1 = k0 + 16 + lm;
        const int key0c = key0 > SEQ - 1 ? SEQ - 1 : key0;
        const int key1c = key1 > SEQ - 1 ? SEQ - 1 : key1;

        // scores: B-layout is 32B contiguous in d -> direct v16bf loads
        v8f s0 = {}, s1 = {};
        {
            const v16bf* kp = reinterpret_cast<const v16bf*>(Kp + (size_t)key0c * HS);
            s0 = wmma_v(qa0.v, kp[lh], s0);        // d = lh*16 + j
            s0 = wmma_v(qa1.v, kp[2 + lh], s0);    // d = 32 + lh*16 + j
        }
        {
            const v16bf* kp = reinterpret_cast<const v16bf*>(Kp + (size_t)key1c * HS);
            s1 = wmma_v(qa0.v, kp[lh], s1);
            s1 = wmma_v(qa1.v, kp[2 + lh], s1);
        }

        // online softmax update (row m = r + 8*lh; 16 lanes per row-half)
        float p0a[8], p1a[8], alpha[8];
#pragma unroll
        for (int r = 0; r < 8; r++) {
            const int qpos = q0 + r + 8 * lh;
            float v0 = (key0 <= qpos && key0 < SEQ) ? s0[r] * scale : NEGV;
            float v1 = (key1 <= qpos && key1 < SEQ) ? s1[r] * scale : NEGV;
            float mx = fmaxf(v0, v1);
#pragma unroll
            for (int off = 1; off < 16; off <<= 1)
                mx = fmaxf(mx, __shfl_xor(mx, off, 32));
            const float mn = fmaxf(mrow[r], mx);
            const float al = __expf(mrow[r] - mn);
            const float p0 = __expf(v0 - mn);
            const float p1 = __expf(v1 - mn);
            float rs = p0 + p1;
#pragma unroll
            for (int off = 1; off < 16; off <<= 1)
                rs += __shfl_xor(rs, off, 32);
            lrow[r] = lrow[r] * al + rs;
            mrow[r] = mn;
            alpha[r] = al; p0a[r] = p0; p1a[r] = p1;
        }
#pragma unroll
        for (int c = 0; c < 4; c++)
#pragma unroll
            for (int r = 0; r < 8; r++) o[c][r] *= alpha[r];

        // stage P (16x32 row-major) and V^T (64x32) in LDS
#pragma unroll
        for (int r = 0; r < 8; r++) {
            const int rl = r + 8 * lh;
            Pl[rl * 32 + lm]      = f32_to_bf16(p0a[r]);
            Pl[rl * 32 + 16 + lm] = f32_to_bf16(p1a[r]);
        }
        {
            int key = k0 + lane; if (key > SEQ - 1) key = SEQ - 1;
            const bf16* vr = Vp + (size_t)key * HS;
            for (int d = 0; d < HS; d++) Vt[d * 32 + lane] = vr[d];
        }
        asm volatile("s_wait_dscnt 0x0" ::: "memory");

        V16 pa;
        {
            const v8bf* pp = reinterpret_cast<const v8bf*>(&Pl[lm * 32]);
            pa.h[0] = pp[2 * lh]; pa.h[1] = pp[2 * lh + 1];
        }
#pragma unroll
        for (int c = 0; c < 4; c++) {
            const v16bf* vp2 =
                reinterpret_cast<const v16bf*>(&Vt[(c * 16 + lm) * 32]);
            o[c] = wmma_v(pa.v, vp2[lh], o[c]);   // key = lh*16 + j
        }
        asm volatile("s_wait_dscnt 0x0" ::: "memory");
    }

    // write: main tokens -> attout (B,T,H*HS); group-0 summary -> attsum
#pragma unroll
    for (int r = 0; r < 8; r++) {
        const int p = q0 + r + 8 * lh;
        if (p >= SEQ) continue;
        const float inv = 1.f / lrow[r];
#pragma unroll
        for (int c = 0; c < 4; c++) {
            const int d = c * 16 + lm;
            const float val = o[c][r] * inv;
            if (p < PGRP) {
                const int t = g * PGRP + p;
                attout[((size_t)(b * TT + t)) * DD + h * HS + d] = f32_to_bf16(val);
            } else if (g == 0) {
                attsum[(size_t)(b * HH + h) * HS + d] = val;
            }
        }
    }
}

// ---------------------------------------------------------------------------
// Output tail: qs, ks (group-0 means), vs (group-0 summary attention output)
// ---------------------------------------------------------------------------
__global__ __launch_bounds__(256)
void finalize_tail(const float* __restrict__ qmean, const float* __restrict__ kmean,
                   const float* __restrict__ attsum, float* __restrict__ outTail) {
    const int i = blockIdx.x * 256 + threadIdx.x; // 0..4095
    if (i >= BB * HH * HS) return;
    const int bh = i >> 6, hs = i & 63;
    outTail[i]                    = qmean[(size_t)(bh * NGRP) * HS + hs];
    outTail[BB * HH * HS + i]     = kmean[(size_t)(bh * NGRP) * HS + hs];
    outTail[2 * BB * HH * HS + i] = attsum[i];
}

// ---------------------------------------------------------------------------
// Host orchestration
// ---------------------------------------------------------------------------
extern "C" void kernel_launch(void* const* d_in, const int* in_sizes, int n_in,
                              void* d_out, int out_size, void* d_ws, size_t ws_size,
                              hipStream_t stream) {
    const float* x     = (const float*)d_in[0];
    const float* Wv    = (const float*)d_in[1];
    const float* Wqk   = (const float*)d_in[2];
    const float* bqk   = (const float*)d_in[3];
    const float* Wproj = (const float*)d_in[4];
    const float* ln1w  = (const float*)d_in[5];
    const float* ln2w  = (const float*)d_in[6];
    const float* w1    = (const float*)d_in[7];
    const float* w2    = (const float*)d_in[8];
    const float* w3    = (const float*)d_in[9];
    float* out = (float*)d_out;

    char* ws = (char*)d_ws;
    size_t off = 0;
    auto alloc = [&](size_t bytes) -> char* {
        char* p = ws + off;
        off += (bytes + 255) & ~(size_t)255;
        return p;
    };

    const size_t M = (size_t)BB * TT;                // 8192
    bf16* WvT    = (bf16*)alloc((size_t)DD * DD * 2);
    bf16* WqkT   = (bf16*)alloc((size_t)2 * DD * DD * 2);
    bf16* WprojT = (bf16*)alloc((size_t)DD * DD * 2);
    bf16* w1T    = (bf16*)alloc((size_t)4 * DD * DD * 2);
    bf16* w3T    = (bf16*)alloc((size_t)4 * DD * DD * 2);
    bf16* w2T    = (bf16*)alloc((size_t)4 * DD * DD * 2);
    bf16* xn     = (bf16*)alloc(M * DD * 2);
    bf16* vbf    = (bf16*)alloc(M * DD * 2);
    bf16* qkbf   = (bf16*)alloc(M * 2 * DD * 2);
    bf16* Qgb    = (bf16*)alloc((size_t)BB * HH * NGRP * SEQ * HS * 2);
    bf16* Kgb    = (bf16*)alloc((size_t)BB * HH * NGRP * SEQ * HS * 2);
    bf16* Vgb    = (bf16*)alloc((size_t)BB * HH * NGRP * SEQ * HS * 2);
    bf16* attb   = (bf16*)alloc(M * DD * 2);
    float* qmean = (float*)alloc((size_t)BB * HH * NGRP * HS * 4);
    float* kmean = (float*)alloc((size_t)BB * HH * NGRP * HS * 4);
    float* attsm = (float*)alloc((size_t)BB * HH * HS * 4);
    float* resf  = (float*)alloc(M * DD * 4);
    bf16* hbf    = (bf16*)alloc(M * DD * 2);
    bf16* actb   = (bf16*)alloc(M * 4 * DD * 2);
    (void)in_sizes; (void)n_in; (void)out_size; (void)ws_size;

    // Weight casts (W[K][N] -> Wt[N][K] bf16)
    transpose_cast<<<dim3(DD / 32, DD / 32), 256, 0, stream>>>(Wv, WvT, DD, DD);
    transpose_cast<<<dim3(2 * DD / 32, DD / 32), 256, 0, stream>>>(Wqk, WqkT, DD, 2 * DD);
    transpose_cast<<<dim3(DD / 32, DD / 32), 256, 0, stream>>>(Wproj, WprojT, DD, DD);
    transpose_cast<<<dim3(4 * DD / 32, DD / 32), 256, 0, stream>>>(w1, w1T, DD, 4 * DD);
    transpose_cast<<<dim3(4 * DD / 32, DD / 32), 256, 0, stream>>>(w3, w3T, DD, 4 * DD);
    transpose_cast<<<dim3(DD / 32, 4 * DD / 32), 256, 0, stream>>>(w2, w2T, 4 * DD, DD);

    // xn = rmsnorm(x, ln1_w)
    rmsnorm_bf16<<<(int)M, 256, 0, stream>>>(x, ln1w, xn);
    // v = xn @ Wv
    gemm_wmma<0><<<dim3(DD / 64, M / 128), 256, 0, stream>>>(
        xn, WvT, nullptr, nullptr, nullptr, nullptr, vbf, (int)M, DD, DD);
    // qk = v @ Wqk + bqk
    gemm_wmma<1><<<dim3(2 * DD / 64, M / 128), 256, 0, stream>>>(
        vbf, WqkT, nullptr, bqk, nullptr, nullptr, qkbf, (int)M, 2 * DD, DD);
    // group + summary-token pack
    pack_qkv<<<BB * HH * NGRP, 256, 0, stream>>>(qkbf, vbf, Qgb, Kgb, Vgb, qmean, kmean);
    // causal attention over S = 1025
    attn_wmma<<<dim3((SEQ + 15) / 16, BB * HH * NGRP), 32, 0, stream>>>(
        Qgb, Kgb, Vgb, attb, attsm);
    // res = x + att @ Wproj
    gemm_wmma<2><<<dim3(DD / 64, M / 128), 256, 0, stream>>>(
        attb, WprojT, nullptr, nullptr, x, resf, nullptr, (int)M, DD, DD);
    // h = rmsnorm(res, ln2_w)
    rmsnorm_bf16<<<(int)M, 256, 0, stream>>>(resf, ln2w, hbf);
    // act = silu(h @ w1) * (h @ w3)
    gemm_wmma<3><<<dim3(4 * DD / 64, M / 128), 256, 0, stream>>>(
        hbf, w1T, w3T, nullptr, nullptr, nullptr, actb, (int)M, 4 * DD, DD);
    // hidden = res + act @ w2  -> d_out
    gemm_wmma<2><<<dim3(DD / 64, M / 128), 256, 0, stream>>>(
        actb, w2T, nullptr, nullptr, resf, out, nullptr, (int)M, DD, 4 * DD);
    // tail outputs (qs, ks, vs)
    finalize_tail<<<(BB * HH * HS + 255) / 256, 256, 0, stream>>>(
        qmean, kmean, attsm, out + M * DD);
}
